// SegmentationHead_841813590056
// MI455X (gfx1250) — matrix operand order repaired and checked by reading, not compiled, not run on hardware
//
#include <hip/hip_runtime.h>

typedef __bf16 bf16x16 __attribute__((ext_vector_type(16)));
typedef float f32x8 __attribute__((ext_vector_type(8)));

#define CIN  256
#define CK   128
#define KTOT 2304   // 256 * 9, K order: k = r*256 + cin  (r = kh*3+kw)
#define NC   21
#define NCHUNK 72   // KTOT / 32

__device__ __forceinline__ unsigned short f2bf(float f) {
  return __builtin_bit_cast(unsigned short, (__bf16)f);   // native v_cvt, RNE
}

union Frag { bf16x16 v; unsigned u[8]; };

// ---------------------------------------------------------------------------
// Kernel 0: repack conv1 weights  w[n][cin][r] (f32)  ->  wbf[n][r*256+cin] (bf16)
// One-time transpose+convert so the GEMM B-stage is contiguous bf16.
// ---------------------------------------------------------------------------
__global__ __launch_bounds__(256) void repack_w(
    const float* __restrict__ w, unsigned short* __restrict__ wbf)
{
  const int n = blockIdx.x;                       // 0..127
  for (int kp = threadIdx.x; kp < KTOT; kp += 256) {
    const int r = kp >> 8, cin = kp & 255;
    wbf[n * KTOT + kp] = f2bf(w[n * KTOT + cin * 9 + r]);
  }
}

// ---------------------------------------------------------------------------
// Kernel 1: conv1 3x3 (256->128) + bias + ReLU as implicit GEMM, bf16 WMMA.
// M=16384, N=128, K=2304 with k = r*256+cin: each 32-chunk has uniform r,
// so halo math is per-chunk; loads are unconditional from clamped addresses
// with a cndmask zero-fill (no exec-mask juggling).
// Ping-pong LDS double buffering, one barrier per K-step.
// ---------------------------------------------------------------------------
__global__ __launch_bounds__(256) void conv1_wmma(
    const float* __restrict__ x, const unsigned short* __restrict__ wbf,
    const float* __restrict__ bias, float* __restrict__ h)
{
  __shared__ unsigned short As[2][128 * 36];   // [m][k], stride 36 bf16
  __shared__ unsigned short Bs[2][128 * 36];   // [n][k]
  const int t = threadIdx.x;
  const int m0 = blockIdx.x * 128;
  const int lane = t & 31, lrow = lane & 15, khalf = lane >> 4;
  const int wv = t >> 5, wm = wv & 3, wn = wv >> 2;

  // Per-thread A-load coordinates (fixed row; 16 contiguous k per thread)
  const int ml = t & 127;
  const int m = m0 + ml;
  const int bimg = m >> 12;
  const int yy = (m >> 6) & 63;
  const int xx = m & 63;
  const int khi = t >> 7;                       // k sub-block: [khi*16, khi*16+16)
  const float* xb = x + (size_t)bimg * CIN * 4096;

  f32x8 acc[2][4] = {};

  // ---- staging lambda: chunk ck -> buffer p ----
  auto stage = [&](int ck, int p) {
    const int k0 = ck << 5;
    const int r = ck >> 3;                      // uniform per chunk
    const int c0 = (ck & 7) << 5;
    const int kh = r / 3, kw = r - kh * 3;      // scalar
    const int iy = yy + kh - 1, ix = xx + kw - 1;
    // clamp to a valid address; zero-fill via select (v_med3 + v_cndmask)
    const int iyc = iy < 0 ? 0 : (iy > 63 ? 63 : iy);
    const int ixc = ix < 0 ? 0 : (ix > 63 ? 63 : ix);
    const bool inb = ((unsigned)iy < 64u) && ((unsigned)ix < 64u);
    const float* ap = xb + (size_t)(c0 + khi * 16) * 4096 + iyc * 64 + ixc;

    unsigned short av[16];
#pragma unroll
    for (int j = 0; j < 16; ++j) {
      float v = ap[j * 4096];                   // unconditional, imm offsets
      v = inb ? v : 0.f;                        // v_cndmask
      av[j] = f2bf(v);
    }
#pragma unroll
    for (int i = 0; i < 8; ++i)
      *(unsigned*)&As[p][ml * 36 + khi * 16 + 2 * i] =
          (unsigned)av[2 * i] | ((unsigned)av[2 * i + 1] << 16);

    // B: coalesced dword loads of pre-converted bf16
#pragma unroll
    for (int j = 0; j < 8; ++j) {
      const int idx = j * 256 + t;
      const int n = idx >> 4, kp = idx & 15;
      const unsigned val = *(const unsigned*)(wbf + n * KTOT + k0 + 2 * kp);
      *(unsigned*)&Bs[p][n * 36 + 2 * kp] = val;
    }
  };

  stage(0, 0);
  __syncthreads();

  int p = 0;
  for (int ck = 0; ck < NCHUNK; ++ck) {
    if (ck + 1 < NCHUNK) stage(ck + 1, p ^ 1);  // overlap with WMMA below

    Frag af[2];
#pragma unroll
    for (int tm = 0; tm < 2; ++tm) {
      const unsigned short* q = &As[p][(wm * 32 + tm * 16 + lrow) * 36];
#pragma unroll
      for (int v = 0; v < 8; ++v) {
        const int kb = khalf * 16 + v * 2;
        af[tm].u[v] = *(const unsigned*)(q + kb);
      }
    }
    Frag bfr[4];
#pragma unroll
    for (int tn = 0; tn < 4; ++tn) {
      const unsigned short* q = &Bs[p][(wn * 64 + tn * 16 + lrow) * 36];
#pragma unroll
      for (int v = 0; v < 8; ++v) {
        const int kb = khalf * 16 + v * 2;
        bfr[tn].u[v] = *(const unsigned*)(q + kb);
      }
    }
#pragma unroll
    for (int tm = 0; tm < 2; ++tm)
#pragma unroll
      for (int tn = 0; tn < 4; ++tn)
        acc[tm][tn] = __builtin_amdgcn_wmma_f32_16x16x32_bf16(
            false, af[tm].v, false, bfr[tn].v, (short)0, acc[tm][tn],
            false, false);

    __syncthreads();
    p ^= 1;
  }

  // Epilogue: bias + ReLU; C layout: lane = N, VGPR r -> M = r + 8*khalf.
#pragma unroll
  for (int tm = 0; tm < 2; ++tm) {
#pragma unroll
    for (int tn = 0; tn < 4; ++tn) {
      const int n = wn * 64 + tn * 16 + lrow;
      const float bv = bias[n];
#pragma unroll
      for (int r = 0; r < 8; ++r) {
        const int mm = m0 + wm * 32 + tm * 16 + khalf * 8 + r;
        float vv = acc[tm][tn][r] + bv;
        h[mm * CK + n] = vv > 0.f ? vv : 0.f;
      }
    }
  }
}

// ---------------------------------------------------------------------------
// Kernel 2: fold base_w into conv2:  Wf[nc,i] = sum_o conv2[nc,o]*base_w[o,i].
// Also emit bf16 conv2 weights, both zero-padded to 32 rows.
// ---------------------------------------------------------------------------
__global__ __launch_bounds__(128) void fuse_weights(
    const float* __restrict__ c2, const float* __restrict__ bw,
    unsigned short* __restrict__ c2bf, unsigned short* __restrict__ wfbf)
{
  const int nc = blockIdx.x;   // 0..31 (>=21 are zero pad)
  const int i = threadIdx.x;   // 0..127
  float acc = 0.f, c2v = 0.f;
  if (nc < NC) {
    c2v = c2[nc * CK + i];
    for (int o = 0; o < CK; ++o)
      acc += c2[nc * CK + o] * bw[o * CK + i];
  }
  c2bf[nc * CK + i] = f2bf(c2v);
  wfbf[nc * CK + i] = f2bf(acc);
}

// ---------------------------------------------------------------------------
// Kernel 3: Wavelet-KAN (DoG) — intrinsically v_exp_f32 bound (268M exps).
// 64 rows/block; a thread owns (o, 32 rows): each strided param load is
// amortized over 32 exps; activation reads are LDS wave-broadcasts.
// Also emits silu(h) bf16 (A matrix for the fused conv2 GEMM).
// ---------------------------------------------------------------------------
__global__ __launch_bounds__(256) void kan_wavelet(
    const float* __restrict__ h, const float* __restrict__ scale,
    const float* __restrict__ trans, const float* __restrict__ wavw,
    unsigned short* __restrict__ wavbf, unsigned short* __restrict__ silubf)
{
  __shared__ float hs[64 * 128];   // 32 KB
  const int t = threadIdx.x;
  const int m0 = blockIdx.x * 64;
#pragma unroll
  for (int j = 0; j < 32; ++j) {
    const int idx = j * 256 + t;                 // row = idx>>7, col = idx&127
    const float v = h[m0 * CK + idx];
    hs[idx] = v;
    const float sg =
        __builtin_amdgcn_rcpf(1.f + __builtin_amdgcn_exp2f(-1.44269504f * v));
    silubf[m0 * CK + idx] = f2bf(v * sg);        // silu(x) = x*sigmoid(x)
  }
  __syncthreads();

  const int o = t & 127;
  const int half = t >> 7;                        // rows [half*32, half*32+32)
  float acc[32];
#pragma unroll
  for (int r = 0; r < 32; ++r) acc[r] = 0.f;

  for (int i = 0; i < CK; ++i) {
    const float sc = scale[o * CK + i];
    const float tr = trans[o * CK + i];
    const float wn = -wavw[o * CK + i];           // fold DoG negation into weight
    const float rs = __builtin_amdgcn_rcpf(sc);
#pragma unroll
    for (int r = 0; r < 32; ++r) {
      const float u = (hs[(half * 32 + r) * CK + i] - tr) * rs;
      acc[r] += wn * u * __builtin_amdgcn_exp2f(-0.72134752f * u * u);
    }
  }
#pragma unroll
  for (int r = 0; r < 32; ++r)
    wavbf[(m0 + half * 32 + r) * CK + o] = f2bf(acc[r]);
}

// ---------------------------------------------------------------------------
// Kernel 4: fused conv2:  out = wav @ conv2^T + silu(h) @ Wf^T + bias.
// Single bf16 WMMA GEMM, M=16384, N=32 (21 used), K=256 (=[wav | silu]).
// Output scattered to NCHW layout.
// ---------------------------------------------------------------------------
__global__ __launch_bounds__(256) void conv2_wmma(
    const unsigned short* __restrict__ wavbf,
    const unsigned short* __restrict__ silubf,
    const unsigned short* __restrict__ c2bf,
    const unsigned short* __restrict__ wfbf,
    const float* __restrict__ bias, float* __restrict__ out)
{
  __shared__ unsigned short Bs[32 * 258];  // [n][k], k<128 -> conv2, else Wf
  const int t = threadIdx.x;
  const int m0 = blockIdx.x * 128;
  const int lane = t & 31, lrow = lane & 15, khalf = lane >> 4;
  const int wv = t >> 5;

#pragma unroll
  for (int j = 0; j < 32; ++j) {
    const int idx = j * 256 + t;
    const int n = idx >> 8;
    const int k = idx & 255;
    Bs[n * 258 + k] = (k < CK) ? c2bf[n * CK + k] : wfbf[n * CK + (k - CK)];
  }
  __syncthreads();

  f32x8 acc[2] = {};
  const int mrow = m0 + wv * 16 + lrow;
#pragma unroll
  for (int s = 0; s < 8; ++s) {
    const int k0 = s * 32;
    const unsigned short* abase =
        (k0 < CK) ? (wavbf + mrow * CK + k0) : (silubf + mrow * CK + (k0 - CK));
    Frag af;
#pragma unroll
    for (int v = 0; v < 8; ++v) {
      const int kb = khalf * 16 + v * 2;
      af.u[v] = *(const unsigned*)(abase + kb);
    }
#pragma unroll
    for (int tn = 0; tn < 2; ++tn) {
      Frag bfr;
#pragma unroll
      for (int v = 0; v < 8; ++v) {
        const int kb = k0 + khalf * 16 + v * 2;
        bfr.u[v] = *(const unsigned*)(&Bs[(tn * 16 + lrow) * 258 + kb]);
      }
      acc[tn] = __builtin_amdgcn_wmma_f32_16x16x32_bf16(
          false, af.v, false, bfr.v, (short)0, acc[tn], false, false);
    }
  }

#pragma unroll
  for (int tn = 0; tn < 2; ++tn) {
    const int n = tn * 16 + lrow;
    if (n < NC) {
      const float bv = bias[n];
#pragma unroll
      for (int r = 0; r < 8; ++r) {
        const int mm = m0 + wv * 16 + khalf * 8 + r;
        const int bimg = mm >> 12;
        out[((bimg * NC + n) << 12) + (mm & 4095)] = acc[tn][r] + bv;
      }
    }
  }
}

// ---------------------------------------------------------------------------
extern "C" void kernel_launch(void* const* d_in, const int* in_sizes, int n_in,
                              void* d_out, int out_size, void* d_ws,
                              size_t ws_size, hipStream_t stream) {
  (void)in_sizes; (void)n_in; (void)out_size; (void)ws_size;
  const float* x      = (const float*)d_in[0];
  const float* c1w    = (const float*)d_in[1];
  const float* c1b    = (const float*)d_in[2];
  const float* kscale = (const float*)d_in[3];
  const float* ktrans = (const float*)d_in[4];
  const float* kwavw  = (const float*)d_in[5];
  const float* kbasew = (const float*)d_in[6];
  const float* c2w    = (const float*)d_in[7];
  const float* c2b    = (const float*)d_in[8];
  float* out = (float*)d_out;

  char* ws = (char*)d_ws;
  float*          h      = (float*)(ws);                                // 8 MiB
  unsigned short* wavbf  = (unsigned short*)(ws + (size_t)(8u  << 20)); // 4 MiB
  unsigned short* silubf = (unsigned short*)(ws + (size_t)(12u << 20)); // 4 MiB
  unsigned short* c2bf   = (unsigned short*)(ws + (size_t)(16u << 20)); // 8 KiB
  unsigned short* wfbf   = (unsigned short*)(ws + (size_t)(16u << 20) + 8192);
  unsigned short* wbf    = (unsigned short*)(ws + (size_t)(16u << 20) + 16384); // 576 KiB

  repack_w    <<<128, 256, 0, stream>>>(c1w, wbf);
  conv1_wmma  <<<128, 256, 0, stream>>>(x, wbf, c1b, h);
  fuse_weights<<<32, 128, 0, stream>>>(c2w, kbasew, c2bf, wfbf);
  kan_wavelet <<<256, 256, 0, stream>>>(h, kscale, ktrans, kwavw, wavbf, silubf);
  conv2_wmma  <<<128, 256, 0, stream>>>(wavbf, silubf, c2bf, wfbf, c2b, out);
}
